// BLinear_33423435498260
// MI455X (gfx1250) — 8-bit matrix operand order repaired and checked by reading, compiled, not run on hardware
//
#include <hip/hip_runtime.h>


typedef int   v16i __attribute__((ext_vector_type(16)));
typedef float v8f  __attribute__((ext_vector_type(8)));
typedef int   v4ii __attribute__((ext_vector_type(4)));

#define M_DIM 8192
#define K_DIM 2048
#define N_DIM 2048
#define TILE  128
#define KB    128
#define LDS_STRIDE 144

#if defined(__gfx1250__) && __has_builtin(__builtin_amdgcn_global_load_async_to_lds_b128)
#define USE_ASYNC_LDS 1
#else
#define USE_ASYNC_LDS 0
#endif

#if USE_ASYNC_LDS
typedef __attribute__((address_space(1))) v4ii* gv4_t;
typedef __attribute__((address_space(3))) v4ii* lv4_t;
#endif

__global__ __launch_bounds__(256)
void sign_to_fp8(const float* __restrict__ x, const float* __restrict__ W,
                 unsigned char* __restrict__ xq, unsigned char* __restrict__ wq) {
  long long t = (long long)blockIdx.x * 256 + threadIdx.x;
  long long i = t * 16;
  const long long nx = (long long)M_DIM * K_DIM;
  const long long nw = (long long)N_DIM * K_DIM;
  const float* src;
  unsigned char* dst;
  if (i < nx) {
    src = x + i;
    dst = xq + i;
  } else {
    i -= nx;
    if (i >= nw) return;
    src = W + i;
    dst = wq + i;
  }
  const float4* s4 = (const float4*)src;
  unsigned w[4];
#pragma unroll
  for (int j = 0; j < 4; ++j) {
    float4 f = s4[j];
    unsigned b0 = f.x > 0.f ? 0x38u : (f.x < 0.f ? 0xB8u : 0x00u);
    unsigned b1 = f.y > 0.f ? 0x38u : (f.y < 0.f ? 0xB8u : 0x00u);
    unsigned b2 = f.z > 0.f ? 0x38u : (f.z < 0.f ? 0xB8u : 0x00u);
    unsigned b3 = f.w > 0.f ? 0x38u : (f.w < 0.f ? 0xB8u : 0x00u);
    w[j] = b0 | (b1 << 8) | (b2 << 16) | (b3 << 24);
  }
  uint4 r;
  r.x = w[0]; r.y = w[1]; r.z = w[2]; r.w = w[3];
  *(uint4*)dst = r;
}

__global__ __launch_bounds__(256)
void bgemm_fp8(const unsigned char* __restrict__ Aq,
               const unsigned char* __restrict__ Bq,
               const float* __restrict__ bias,
               float* __restrict__ out) {
  __shared__ __align__(16) unsigned char ldsA[2][TILE * LDS_STRIDE];
  __shared__ __align__(16) unsigned char ldsB[2][TILE * LDS_STRIDE];

  const int tid  = threadIdx.x;
  const int wave = tid >> 5;
  const int lane = tid & 31;
  const int half = lane >> 4;
  const int l16  = lane & 15;

  const int row0 = blockIdx.y * TILE;
  const int col0 = blockIdx.x * TILE;
  const int wr   = (wave >> 1) * 32;
  const int wc   = (wave & 1) * 64;

  v8f acc[2][4] = {};

  const int ldr = tid >> 3;
  const int ldc = (tid & 7) * 16;

  auto stage = [&](int buf, int kb) {
#pragma unroll
    for (int it = 0; it < 4; ++it) {
      int r = ldr + it * 32;
      const unsigned char* ga = Aq + (long long)(row0 + r) * K_DIM + kb + ldc;
      const unsigned char* gb = Bq + (long long)(col0 + r) * K_DIM + kb + ldc;
      unsigned char* la = &ldsA[buf][r * LDS_STRIDE + ldc];
      unsigned char* lb = &ldsB[buf][r * LDS_STRIDE + ldc];
#if USE_ASYNC_LDS
      __builtin_amdgcn_global_load_async_to_lds_b128((gv4_t)ga, (lv4_t)la, 0, 0);
      __builtin_amdgcn_global_load_async_to_lds_b128((gv4_t)gb, (lv4_t)lb, 0, 0);
#else
      *(uint4*)la = *(const uint4*)ga;
      *(uint4*)lb = *(const uint4*)gb;
#endif
    }
  };

  stage(0, 0);
#if USE_ASYNC_LDS
  __builtin_amdgcn_s_wait_asynccnt(0);
#endif
  __syncthreads();

  for (int kb = 0; kb < K_DIM; kb += KB) {
    const int cur = (kb / KB) & 1;

    if (kb + KB < K_DIM) stage(cur ^ 1, kb + KB);

    v16i afrag[2];
#pragma unroll
    for (int ti = 0; ti < 2; ++ti) {
      const unsigned char* base =
          &ldsA[cur][(wr + ti * 16 + l16) * LDS_STRIDE] + half * 8;
      unsigned long long* d = (unsigned long long*)&afrag[ti];
#pragma unroll
      for (int j = 0; j < 4; ++j) {
        d[j]     = *(const unsigned long long*)(base + j * 16);
        d[j + 4] = *(const unsigned long long*)(base + 64 + j * 16);
      }
    }

    v16i bfrag[4];
#pragma unroll
    for (int tj = 0; tj < 4; ++tj) {
      const unsigned char* base =
          &ldsB[cur][(wc + tj * 16 + l16) * LDS_STRIDE] + half * 8;
      unsigned long long* d = (unsigned long long*)&bfrag[tj];
#pragma unroll
      for (int j = 0; j < 4; ++j) {
        d[j]     = *(const unsigned long long*)(base + j * 16);
        d[j + 4] = *(const unsigned long long*)(base + 64 + j * 16);
      }
    }

#pragma unroll
    for (int ti = 0; ti < 2; ++ti)
#pragma unroll
      for (int tj = 0; tj < 4; ++tj)
        acc[ti][tj] = __builtin_amdgcn_wmma_f32_16x16x128_fp8_fp8(
            afrag[ti], bfrag[tj], (short)0, acc[ti][tj],
             false,  false);

#if USE_ASYNC_LDS
    __builtin_amdgcn_s_wait_asynccnt(0);
#endif
    __syncthreads();
  }

#pragma unroll
  for (int tj = 0; tj < 4; ++tj) {
    int col = col0 + wc + tj * 16 + l16;
    float bv = bias[col];
#pragma unroll
    for (int ti = 0; ti < 2; ++ti) {
#pragma unroll
      for (int v = 0; v < 8; ++v) {
        int row = row0 + wr + ti * 16 + half * 8 + v;
        out[(long long)row * N_DIM + col] = acc[ti][tj][v] + bv;
      }
    }
  }
}

extern "C" void kernel_launch(void* const* d_in, const int* in_sizes, int n_in,
                              void* d_out, int out_size, void* d_ws, size_t ws_size,
                              hipStream_t stream) {
  const float* x = (const float*)d_in[0];
  const float* W = (const float*)d_in[1];
  const float* b = (const float*)d_in[2];
  float* out = (float*)d_out;

  unsigned char* xq = (unsigned char*)d_ws;
  unsigned char* wq = xq + (size_t)M_DIM * K_DIM;

  const long long total16 =
      ((long long)M_DIM * K_DIM + (long long)N_DIM * K_DIM) / 16;
  const int pack_blocks = (int)((total16 + 255) / 256);
  sign_to_fp8<<<pack_blocks, 256, 0, stream>>>(x, W, xq, wq);

  dim3 grid(N_DIM / TILE, M_DIM / TILE);
  bgemm_fp8<<<grid, 256, 0, stream>>>(xq, wq, b, out);
}
